// GATLayer_82197084111203
// MI455X (gfx1250) — matrix operand order repaired and checked by reading, not compiled, not run on hardware
//
#include <hip/hip_runtime.h>
#include <hip/hip_bf16.h>

typedef __attribute__((ext_vector_type(16))) _Float16 v16h;
typedef __attribute__((ext_vector_type(8)))  _Float16 v8h;
typedef __attribute__((ext_vector_type(8)))  float    v8f;

// Problem constants (from the reference)
constexpr int   NN   = 50000;
constexpr int   EE   = 800000;
constexpr int   FIN  = 256;
constexpr int   HH   = 4;
constexpr int   CC   = 32;
constexpr int   HC   = HH * CC;           // 128
constexpr int   MEDG = EE + NN;           // edges incl. self loops
constexpr int   MTILES = NN / 16;         // 3125 exactly
constexpr float SLOPE_ATT = 0.2f;
constexpr float SLOPE_ACT = 0.01f;
constexpr float BN_EPS    = 1e-5f;

// ---------------------------------------------------------------------------
// f32 -> f16 conversion (x and W)
__global__ void cvt_f16_kernel(const float* __restrict__ src,
                               _Float16* __restrict__ dst, size_t n) {
    size_t i = (size_t)blockIdx.x * blockDim.x + threadIdx.x;
    if (i < n) dst[i] = (_Float16)src[i];
}

// ---------------------------------------------------------------------------
// h = x @ W^T via WMMA f32 = f16*f16.
//  * W (64 KB) is staged once per workgroup into LDS with the CDNA5 async
//    global->LDS path (global_load_async_to_lds_b128, s_wait_asynccnt).
//  * One wave per 16-row strip computes all 8 column tiles: A streamed from
//    HBM exactly once, B fragments served from LDS (ds_load_b128).
//  * h stored as f16 to halve the downstream per-edge gather traffic.
__global__ __launch_bounds__(256) void gemm_wmma_kernel(
        const _Float16* __restrict__ xb,   // [N, 256] f16
        const _Float16* __restrict__ wb,   // [128, 256] f16 (row j = output col)
        _Float16* __restrict__ hout) {     // [N, 128] f16
    __shared__ _Float16 wlds[HC * FIN];    // 64 KB

    // ---- cooperative async copy of W into LDS: 256 threads x 16 x b128 ----
    {
        const unsigned t = threadIdx.x;
        const char* gsrc = (const char*)wb;
        unsigned ldsBase = (unsigned)(uintptr_t)(&wlds[0]);  // addr[31:0] = LDS offset
        #pragma unroll
        for (int i = 0; i < 16; ++i) {
            unsigned byteOff = (unsigned)(i * 256 + t) * 16u;  // coalesced 16B chunks
            unsigned laddr = ldsBase + byteOff;
            const void* gaddr = gsrc + byteOff;
            asm volatile("global_load_async_to_lds_b128 %0, %1, off"
                         :: "v"(laddr), "v"(gaddr) : "memory");
        }
        asm volatile("s_wait_asynccnt 0" ::: "memory");
    }
    __syncthreads();

    const int waveId  = blockIdx.x * 8 + (threadIdx.x >> 5);
    if (waveId < MTILES) {                 // wave-uniform guard: EXEC all-ones inside
        const int lane    = threadIdx.x & 31;
        const int laneLow = lane & 15;
        const int laneHi  = lane >> 4;

        const _Float16* aRow = xb + (size_t)(waveId * 16 + laneLow) * FIN;

        v8f acc[8];
        #pragma unroll
        for (int t = 0; t < 8; ++t) acc[t] = (v8f){0.f,0.f,0.f,0.f,0.f,0.f,0.f,0.f};

        #pragma unroll
        for (int kc = 0; kc < 8; ++kc) {
            const int kb = kc * 32;
            // A fragment (16x32 f16): lane<16 holds K {0..7,16..23}, lane>=16 {8..15,24..31}
            v8h alo = *(const v8h*)(aRow + kb + laneHi * 8);
            v8h ahi = *(const v8h*)(aRow + kb + 16 + laneHi * 8);
            v16h a;
            #pragma unroll
            for (int i = 0; i < 8; ++i) { a[i] = alo[i]; a[i + 8] = ahi[i]; }

            #pragma unroll
            for (int nt = 0; nt < 8; ++nt) {
                // B fragment (32x16) from LDS: lane = output col, lane<16 K {0..7,16..23}, lane>=16 {8..15,24..31}
                v8h blo = *(const v8h*)(&wlds[(size_t)(nt * 16 + laneLow) * FIN
                                              + kb + laneHi * 8]);
                v8h bhi = *(const v8h*)(&wlds[(size_t)(nt * 16 + laneLow) * FIN
                                              + kb + 16 + laneHi * 8]);
                v16h b;
                #pragma unroll
                for (int i = 0; i < 8; ++i) { b[i] = blo[i]; b[i + 8] = bhi[i]; }
                acc[nt] = __builtin_amdgcn_wmma_f32_16x16x32_f16(
                    false, a, false, b, (short)0, acc[nt], false, false);
            }
        }
        // C/D layout: VGPR r -> M = r + laneHi*8, N = laneLow
        #pragma unroll
        for (int nt = 0; nt < 8; ++nt) {
            _Float16* op = hout + (size_t)(waveId * 16 + laneHi * 8) * HC
                                + nt * 16 + laneLow;
            #pragma unroll
            for (int r = 0; r < 8; ++r) op[(size_t)r * HC] = (_Float16)acc[nt][r];
        }
    }
}

// ---------------------------------------------------------------------------
// Per-node attention scores: a_s[n,h] = <h[n,h,:], att_src[h,:]>, same for dst.
__global__ void scores_kernel(const _Float16* __restrict__ h16,
                              const float* __restrict__ att_src,
                              const float* __restrict__ att_dst,
                              float* __restrict__ a_s,
                              float* __restrict__ a_d) {
    int idx = blockIdx.x * blockDim.x + threadIdx.x;   // node*H + head
    if (idx >= NN * HH) return;
    int node = idx >> 2, hd = idx & 3;
    const _Float16* hp = h16 + (size_t)node * HC + hd * CC;
    const float* as = att_src + hd * CC;
    const float* ad = att_dst + hd * CC;
    float s = 0.f, d = 0.f;
    #pragma unroll
    for (int c = 0; c < CC; ++c) {
        float v = (float)hp[c];
        s += v * as[c];
        d += v * ad[c];
    }
    a_s[idx] = s;
    a_d[idx] = d;
}

// ---------------------------------------------------------------------------
__device__ __forceinline__ void edge_nodes(const int* __restrict__ ei,
                                           int e, int& s, int& d) {
    if (e < EE) { s = ei[e]; d = ei[EE + e]; }
    else        { s = d = e - EE; }                    // self loops
}

__device__ __forceinline__ float leaky_att(float v) {
    return v > 0.f ? v : SLOPE_ATT * v;
}

// Ordered-int float atomic max (addr must be initialized to -inf)
__device__ __forceinline__ void atomicMaxF(float* addr, float val) {
    if (val >= 0.f) atomicMax((int*)addr, __float_as_int(val));
    else            atomicMin((unsigned int*)addr, __float_as_uint(val));
}

// init: e_max = -inf, denom/sums/sumsq = 0
__global__ void init_kernel(float* __restrict__ e_max, float* __restrict__ denom,
                            float* __restrict__ sums, float* __restrict__ sumsq) {
    int i = blockIdx.x * blockDim.x + threadIdx.x;
    if (i < NN * HH) { e_max[i] = -__builtin_inff(); denom[i] = 0.f; }
    if (i < HC)      { sums[i] = 0.f; sumsq[i] = 0.f; }
}

// segment max of edge logits per (dst, head)
__global__ void att_max_kernel(const int* __restrict__ ei,
                               const float* __restrict__ a_s,
                               const float* __restrict__ a_d,
                               float* __restrict__ e_max) {
    int e = blockIdx.x * blockDim.x + threadIdx.x;
    if (e >= MEDG) return;
    int s, d; edge_nodes(ei, e, s, d);
    float4 as = *(const float4*)(a_s + (size_t)s * HH);
    float4 ad = *(const float4*)(a_d + (size_t)d * HH);
    atomicMaxF(&e_max[(size_t)d * HH + 0], leaky_att(as.x + ad.x));
    atomicMaxF(&e_max[(size_t)d * HH + 1], leaky_att(as.y + ad.y));
    atomicMaxF(&e_max[(size_t)d * HH + 2], leaky_att(as.z + ad.z));
    atomicMaxF(&e_max[(size_t)d * HH + 3], leaky_att(as.w + ad.w));
}

// segment sum of exp(logit - max) per (dst, head)
__global__ void att_sum_kernel(const int* __restrict__ ei,
                               const float* __restrict__ a_s,
                               const float* __restrict__ a_d,
                               const float* __restrict__ e_max,
                               float* __restrict__ denom) {
    int e = blockIdx.x * blockDim.x + threadIdx.x;
    if (e >= MEDG) return;
    int s, d; edge_nodes(ei, e, s, d);
    float4 as = *(const float4*)(a_s + (size_t)s * HH);
    float4 ad = *(const float4*)(a_d + (size_t)d * HH);
    const float* mx = e_max + (size_t)d * HH;
    atomicAdd(&denom[(size_t)d * HH + 0], __expf(leaky_att(as.x + ad.x) - mx[0]));
    atomicAdd(&denom[(size_t)d * HH + 1], __expf(leaky_att(as.y + ad.y) - mx[1]));
    atomicAdd(&denom[(size_t)d * HH + 2], __expf(leaky_att(as.z + ad.z) - mx[2]));
    atomicAdd(&denom[(size_t)d * HH + 3], __expf(leaky_att(as.w + ad.w) - mx[3]));
}

// out[dst,f] += alpha(e,head(f)) * h[src,f]; 128 threads per edge, 2 edges/block
__global__ __launch_bounds__(256) void aggregate_kernel(
        const int* __restrict__ ei,
        const float* __restrict__ a_s,
        const float* __restrict__ a_d,
        const float* __restrict__ e_max,
        const float* __restrict__ denom,
        const _Float16* __restrict__ h16,
        float* __restrict__ out) {
    int eid = blockIdx.x * 2 + (threadIdx.x >> 7);
    if (eid >= MEDG) return;
    int f  = threadIdx.x & 127;
    int hd = f >> 5;
    int s, d; edge_nodes(ei, eid, s, d);
    float v  = leaky_att(a_s[(size_t)s * HH + hd] + a_d[(size_t)d * HH + hd]);
    float w  = __expf(v - e_max[(size_t)d * HH + hd]) / denom[(size_t)d * HH + hd];
    float hv = (float)h16[(size_t)s * HC + f];
    atomicAdd(&out[(size_t)d * HC + f], w * hv);
}

// ---------------------------------------------------------------------------
// per-feature batch statistics of (out + bias)
__global__ __launch_bounds__(128) void stats_kernel(
        const float* __restrict__ out, const float* __restrict__ bias,
        float* __restrict__ sums, float* __restrict__ sumsq) {
    int j = threadIdx.x;                   // 0..127 feature
    float b = bias[j];
    float s = 0.f, ss = 0.f;
    for (int r = blockIdx.x; r < NN; r += gridDim.x) {
        float v = out[(size_t)r * HC + j] + b;
        s += v; ss += v * v;
    }
    atomicAdd(&sums[j], s);
    atomicAdd(&sumsq[j], ss);
}

__global__ __launch_bounds__(128) void bnparam_kernel(
        const float* __restrict__ sums, const float* __restrict__ sumsq,
        const float* __restrict__ gamma, const float* __restrict__ beta,
        float* __restrict__ bnA, float* __restrict__ bnB) {
    int j = threadIdx.x;
    float mu  = sums[j] / (float)NN;
    float var = sumsq[j] / (float)NN - mu * mu;
    float rs  = rsqrtf(var + BN_EPS);
    float a   = rs * gamma[j];
    bnA[j] = a;
    bnB[j] = beta[j] - mu * a;
}

// in-place: y = leakyrelu_0.01( (out+bias)*A + B )
__global__ void finalize_kernel(float* __restrict__ out,
                                const float* __restrict__ bias,
                                const float* __restrict__ bnA,
                                const float* __restrict__ bnB) {
    size_t i = (size_t)blockIdx.x * blockDim.x + threadIdx.x;
    if (i >= (size_t)NN * HC) return;
    int j = (int)(i & (HC - 1));
    float v = (out[i] + bias[j]) * bnA[j] + bnB[j];
    out[i] = v > 0.f ? v : SLOPE_ACT * v;
}

// ---------------------------------------------------------------------------
extern "C" void kernel_launch(void* const* d_in, const int* in_sizes, int n_in,
                              void* d_out, int out_size, void* d_ws, size_t ws_size,
                              hipStream_t stream) {
    const float* x        = (const float*)d_in[0];
    const int*   edge_idx = (const int*)  d_in[1];
    const float* W        = (const float*)d_in[2];
    const float* att_src  = (const float*)d_in[3];
    const float* att_dst  = (const float*)d_in[4];
    const float* bias     = (const float*)d_in[5];
    const float* gamma    = (const float*)d_in[6];
    const float* beta     = (const float*)d_in[7];
    float* out = (float*)d_out;

    // workspace carve-up (all offsets 256B aligned)
    char* ws = (char*)d_ws;
    size_t off = 0;
    _Float16* xb   = (_Float16*)(ws + off); off += (size_t)NN * FIN * 2;   // 25.6 MB
    _Float16* wb   = (_Float16*)(ws + off); off += (size_t)HC * FIN * 2;   // 64 KB
    _Float16* h16  = (_Float16*)(ws + off); off += (size_t)NN * HC  * 2;   // 12.8 MB
    float*    a_s  = (float*)   (ws + off); off += (size_t)NN * HH  * 4;
    float*    a_d  = (float*)   (ws + off); off += (size_t)NN * HH  * 4;
    float*    emax = (float*)   (ws + off); off += (size_t)NN * HH  * 4;
    float*    den  = (float*)   (ws + off); off += (size_t)NN * HH  * 4;
    float*    sums = (float*)   (ws + off); off += 512;
    float*    sumq = (float*)   (ws + off); off += 512;
    float*    bnA  = (float*)   (ws + off); off += 512;
    float*    bnB  = (float*)   (ws + off); off += 512;
    (void)ws_size; (void)n_in; (void)in_sizes; (void)out_size;

    // zero the accumulation buffer (graph-capturable)
    hipMemsetAsync(d_out, 0, (size_t)NN * HC * sizeof(float), stream);

    // f32 -> f16
    {
        size_t nx = (size_t)NN * FIN;
        cvt_f16_kernel<<<(unsigned)((nx + 255) / 256), 256, 0, stream>>>(x, xb, nx);
        size_t nw = (size_t)HC * FIN;
        cvt_f16_kernel<<<(unsigned)((nw + 255) / 256), 256, 0, stream>>>(W, wb, nw);
    }

    // WMMA GEMM: 3125 strips, 8 waves/block, W staged in LDS via async copy
    gemm_wmma_kernel<<<(MTILES + 7) / 8, 256, 0, stream>>>(xb, wb, h16);

    // per-node scores
    scores_kernel<<<(NN * HH + 255) / 256, 256, 0, stream>>>(h16, att_src, att_dst, a_s, a_d);

    // init softmax state
    init_kernel<<<(NN * HH + 255) / 256, 256, 0, stream>>>(emax, den, sums, sumq);

    // segment softmax over edges
    att_max_kernel<<<(MEDG + 255) / 256, 256, 0, stream>>>(edge_idx, a_s, a_d, emax);
    att_sum_kernel<<<(MEDG + 255) / 256, 256, 0, stream>>>(edge_idx, a_s, a_d, emax, den);

    // scatter-add aggregation (2 edges per 256-thread block)
    aggregate_kernel<<<(MEDG + 1) / 2, 256, 0, stream>>>(edge_idx, a_s, a_d, emax, den, h16, out);

    // BatchNorm stats + params + fused bias/BN/LeakyReLU
    stats_kernel<<<512, 128, 0, stream>>>(out, bias, sums, sumq);
    bnparam_kernel<<<1, 128, 0, stream>>>(sums, sumq, gamma, beta, bnA, bnB);
    finalize_kernel<<<(int)(((size_t)NN * HC + 255) / 256), 256, 0, stream>>>(out, bias, bnA, bnB);
}
